// UnsupervisedGAT_32538672234907
// MI455X (gfx1250) — matrix operand order repaired and checked by reading, not compiled, not run on hardware
//
#include <hip/hip_runtime.h>

typedef __attribute__((ext_vector_type(16))) __bf16 v16bf;
typedef __attribute__((ext_vector_type(8)))  __bf16 v8bf;
typedef __attribute__((ext_vector_type(8)))  float  v8f;

#define ATTN_SLOPE 0.2f
#define ACT_SLOPE  0.01f
#define TILES_PER_WAVE 8

__device__ __forceinline__ float lrelu(float x, float s) { return x > 0.f ? x : x * s; }

// order-preserving float<->uint encoding for atomicMax on floats (incl. negatives)
__device__ __forceinline__ unsigned enc_f(float f) {
    unsigned u = __float_as_uint(f);
    return (u & 0x80000000u) ? ~u : (u | 0x80000000u);
}
__device__ __forceinline__ float dec_f(unsigned u) {
    return (u & 0x80000000u) ? __uint_as_float(u ^ 0x80000000u) : __uint_as_float(~u);
}

// ---------- W (f32 64x64, [k][n]) -> bf16 transposed [n][k] ----------
__global__ void k_convw(const float* __restrict__ W, __bf16* __restrict__ WbT) {
    int t = blockIdx.x * blockDim.x + threadIdx.x;
    if (t < 64 * 64) {
        int n = t >> 6, k = t & 63;
        WbT[t] = (__bf16)W[k * 64 + n];
    }
}

// ---------------- Projection GEMM via WMMA -------------
// One wave per block; B (64x64 weights) held in registers; each wave computes
// TILES_PER_WAVE row-tiles of 16 rows. X:[N,64] f32, WbT:[64(n)][64(k)] bf16,
// hout:[N,64] f32. Optional leaky-relu applied to X on load (layer-2 input).
__global__ __launch_bounds__(32) void k_gemm_wmma(const float* __restrict__ x,
                                                  const __bf16* __restrict__ WbT,
                                                  float* __restrict__ hout,
                                                  int apply_act, int ntiles) {
    const int lane = threadIdx.x;
    const int half = lane >> 4;     // 0: lanes 0-15, 1: lanes 16-31
    const int l15  = lane & 15;

    // ---- B resident in registers. Layout (ISA 7.12.2): lane group `half`
    // holds K = e + (e>=8?8:0) + half*8 for element e, column N = l15. With WbT[n][k]
    // the two 8-element runs are contiguous -> two b128 loads per vector. ----
    v16bf b[4][2];
#pragma unroll
    for (int c = 0; c < 4; ++c) {
        const __bf16* wp = WbT + (c * 16 + l15) * 64 + half * 8;
#pragma unroll
        for (int kb = 0; kb < 2; ++kb) {
            v8bf lo = *(const v8bf*)(wp + kb * 32);
            v8bf hi = *(const v8bf*)(wp + kb * 32 + 16);
            b[c][kb] = __builtin_shufflevector(lo, hi, 0, 1, 2, 3, 4, 5, 6, 7,
                                               8, 9, 10, 11, 12, 13, 14, 15);
        }
    }

    for (int t = 0; t < TILES_PER_WAVE; ++t) {
        const int tile = blockIdx.x * TILES_PER_WAVE + t;   // uniform across wave
        if (tile >= ntiles) break;
        const int row0 = tile * 16;
        const int arow = row0 + l15;

        // ---- A tiles: element e -> K = e + (e>=8?8:0) + half*8; the two
        // 8-element runs are contiguous in memory -> float4 loads. ----
        const float4* xp = (const float4*)(x + arow * 64);
        const int o = half * 2;                // float4 index of K = half*8
        float a0f[16], a1f[16];
        *(float4*)&a0f[0]  = xp[o];      *(float4*)&a0f[4]  = xp[o + 1];
        *(float4*)&a0f[8]  = xp[o + 4];  *(float4*)&a0f[12] = xp[o + 5];
        *(float4*)&a1f[0]  = xp[o + 8];  *(float4*)&a1f[4]  = xp[o + 9];
        *(float4*)&a1f[8]  = xp[o + 12]; *(float4*)&a1f[12] = xp[o + 13];

        v16bf a0, a1;
#pragma unroll
        for (int e = 0; e < 16; ++e) {
            float f0 = a0f[e], f1 = a1f[e];
            if (apply_act) { f0 = lrelu(f0, ACT_SLOPE); f1 = lrelu(f1, ACT_SLOPE); }
            a0[e] = (__bf16)f0;
            a1[e] = (__bf16)f1;
        }

#pragma unroll
        for (int c = 0; c < 4; ++c) {
            v8f acc = {};
            acc = __builtin_amdgcn_wmma_f32_16x16x32_bf16(false, a0, false, b[c][0],
                                                          (short)0, acc, false, false);
            acc = __builtin_amdgcn_wmma_f32_16x16x32_bf16(false, a1, false, b[c][1],
                                                          (short)0, acc, false, false);
            // D layout: VGPR j -> M = half*8 + j, N = l15
            const int mbase = half * 8;
#pragma unroll
            for (int j = 0; j < 8; ++j)
                hout[(row0 + mbase + j) * 64 + c * 16 + l15] = acc[j];
        }
    }
}

// ---------------- per-node attention scalars el/er ------
__global__ void k_elr(const float* __restrict__ hbuf, const float* __restrict__ al,
                      const float* __restrict__ ar, float* __restrict__ el,
                      float* __restrict__ er, int Nn) {
    int t = blockIdx.x * blockDim.x + threadIdx.x;
    if (t >= Nn * 4) return;
    int n = t >> 2, h = t & 3;
    const float* hr = hbuf + n * 64 + h * 16;
    const float* a = al + h * 16;
    const float* b = ar + h * 16;
    float sl = 0.f, sr = 0.f;
#pragma unroll
    for (int f = 0; f < 16; ++f) { float v = hr[f]; sl += v * a[f]; sr += v * b[f]; }
    el[t] = sl;
    er[t] = sr;
}

// ---------------- init: out=bias, m=-inf, s=0 -----------
__global__ void k_init(float* __restrict__ outb, const float* __restrict__ bias,
                       unsigned* __restrict__ m, float* __restrict__ ssum, int Nn) {
    int t = blockIdx.x * blockDim.x + threadIdx.x;
    if (t >= Nn * 64) return;
    int d = t & 63;
    outb[t] = bias[d];
    if (d < 4) {
        int n = t >> 6;
        m[n * 4 + d] = 0x007FFFFFu;   // enc(-inf)
        ssum[n * 4 + d] = 0.f;
    }
}

// ---------------- segment max over incoming edges -------
__global__ void k_edge_max(const int* __restrict__ src, const int* __restrict__ dst,
                           const float* __restrict__ el, const float* __restrict__ er,
                           unsigned* __restrict__ m, int Ee) {
    int e = blockIdx.x * blockDim.x + threadIdx.x;
    if (e >= Ee) return;
    int s_ = src[e], d_ = dst[e];
#pragma unroll
    for (int h = 0; h < 4; ++h) {
        float v = lrelu(el[s_ * 4 + h] + er[d_ * 4 + h], ATTN_SLOPE);
        atomicMax(&m[d_ * 4 + h], enc_f(v));
    }
}

// ---------------- segment sum of exp(logit - max) -------
__global__ void k_edge_sum(const int* __restrict__ src, const int* __restrict__ dst,
                           const float* __restrict__ el, const float* __restrict__ er,
                           const unsigned* __restrict__ m, float* __restrict__ ssum, int Ee) {
    int e = blockIdx.x * blockDim.x + threadIdx.x;
    if (e >= Ee) return;
    int s_ = src[e], d_ = dst[e];
#pragma unroll
    for (int h = 0; h < 4; ++h) {
        float v = lrelu(el[s_ * 4 + h] + er[d_ * 4 + h], ATTN_SLOPE);
        float ex = __expf(v - dec_f(m[d_ * 4 + h]));
        atomicAdd(&ssum[d_ * 4 + h], ex);
    }
}

// ---------------- weighted message aggregation ----------
// 16 threads per edge: thread sub = head*4 + fgroup, each does 4 features.
__global__ void k_edge_agg(const int* __restrict__ src, const int* __restrict__ dst,
                           const float* __restrict__ el, const float* __restrict__ er,
                           const unsigned* __restrict__ m, const float* __restrict__ ssum,
                           const float* __restrict__ hbuf, float* __restrict__ outb,
                           int Ee) {
    long long t = (long long)blockIdx.x * blockDim.x + threadIdx.x;
    if (t >= (long long)Ee * 16) return;
    int e   = (int)(t >> 4);
    int sub = (int)(t & 15);
    int h   = sub >> 2;
    int fg  = (sub & 3) * 4;
    int s_ = src[e], d_ = dst[e];
    float logit = lrelu(el[s_ * 4 + h] + er[d_ * 4 + h], ATTN_SLOPE);
    float alpha = __expf(logit - dec_f(m[d_ * 4 + h])) / ssum[d_ * 4 + h];
    const float4 hv = *(const float4*)(hbuf + s_ * 64 + h * 16 + fg);
    float* ob = outb + d_ * 64 + h * 16 + fg;
    atomicAdd(ob + 0, alpha * hv.x);
    atomicAdd(ob + 1, alpha * hv.y);
    atomicAdd(ob + 2, alpha * hv.z);
    atomicAdd(ob + 3, alpha * hv.w);
}

// ---------------- mean pooling readout ------------------
__global__ void k_pool_init(float* __restrict__ out, float* __restrict__ counts) {
    int t = blockIdx.x * blockDim.x + threadIdx.x;
    if (t < 256 * 64) out[t] = 0.f;
    if (t < 256) counts[t] = 0.f;
}
__global__ void k_pool(const float* __restrict__ x2, const int* __restrict__ gid,
                       float* __restrict__ out, float* __restrict__ counts, int Nn) {
    int t = blockIdx.x * blockDim.x + threadIdx.x;
    if (t >= Nn * 64) return;
    int n = t >> 6, d = t & 63;
    int g = gid[n];
    atomicAdd(&out[g * 64 + d], x2[t]);
    if (d == 0) atomicAdd(&counts[g], 1.f);
}
__global__ void k_pool_norm(float* __restrict__ out, const float* __restrict__ counts) {
    int t = blockIdx.x * blockDim.x + threadIdx.x;
    if (t < 256 * 64) out[t] /= fmaxf(counts[t >> 6], 1.f);
}

extern "C" void kernel_launch(void* const* d_in, const int* in_sizes, int n_in,
                              void* d_out, int out_size, void* d_ws, size_t ws_size,
                              hipStream_t stream) {
    const float* feats = (const float*)d_in[0];
    const int*   src   = (const int*)d_in[1];
    const int*   dst   = (const int*)d_in[2];
    const int*   gid   = (const int*)d_in[3];
    const float* Wp[2]  = {(const float*)d_in[4], (const float*)d_in[8]};
    const float* alp[2] = {(const float*)d_in[5], (const float*)d_in[9]};
    const float* arp[2] = {(const float*)d_in[6], (const float*)d_in[10]};
    const float* bp[2]  = {(const float*)d_in[7], (const float*)d_in[11]};

    const int Nn = in_sizes[3];      // 50000
    const int Ee = in_sizes[1];      // 1600000
    float* out = (float*)d_out;

    // workspace layout
    char* ws = (char*)d_ws;
    size_t off = 0;
    float*    buf_h  = (float*)(ws + off);  off += (size_t)Nn * 64 * 4;   // h per layer
    float*    buf_o  = (float*)(ws + off);  off += (size_t)Nn * 64 * 4;   // layer output
    float*    el     = (float*)(ws + off);  off += (size_t)Nn * 4 * 4;
    float*    er     = (float*)(ws + off);  off += (size_t)Nn * 4 * 4;
    unsigned* mmax   = (unsigned*)(ws + off); off += (size_t)Nn * 4 * 4;
    float*    ssum   = (float*)(ws + off);  off += (size_t)Nn * 4 * 4;
    __bf16*   WbT    = (__bf16*)(ws + off); off += 64 * 64 * 2;
    float*    counts = (float*)(ws + off);  off += 256 * 4;
    (void)ws_size; (void)n_in; (void)out_size;

    const int TB = 256;
    const int ntiles = Nn / 16;                                   // 3125
    const int gemm_blocks = (ntiles + TILES_PER_WAVE - 1) / TILES_PER_WAVE;

    for (int L = 0; L < 2; ++L) {
        const float* xin = (L == 0) ? feats : buf_o;
        k_convw<<<(64 * 64 + TB - 1) / TB, TB, 0, stream>>>(Wp[L], WbT);
        k_gemm_wmma<<<gemm_blocks, 32, 0, stream>>>(xin, WbT, buf_h, L, ntiles);
        k_elr<<<(Nn * 4 + TB - 1) / TB, TB, 0, stream>>>(buf_h, alp[L], arp[L], el, er, Nn);
        k_init<<<(Nn * 64 + TB - 1) / TB, TB, 0, stream>>>(buf_o, bp[L], mmax, ssum, Nn);
        k_edge_max<<<(Ee + TB - 1) / TB, TB, 0, stream>>>(src, dst, el, er, mmax, Ee);
        k_edge_sum<<<(Ee + TB - 1) / TB, TB, 0, stream>>>(src, dst, el, er, mmax, ssum, Ee);
        long long aggT = (long long)Ee * 16;
        k_edge_agg<<<(int)((aggT + TB - 1) / TB), TB, 0, stream>>>(src, dst, el, er, mmax,
                                                                   ssum, buf_h, buf_o, Ee);
    }
    // mean pooling into d_out
    k_pool_init<<<(256 * 64 + TB - 1) / TB, TB, 0, stream>>>(out, counts);
    k_pool<<<(Nn * 64 + TB - 1) / TB, TB, 0, stream>>>(buf_o, gid, out, counts, Nn);
    k_pool_norm<<<(256 * 64 + TB - 1) / TB, TB, 0, stream>>>(out, counts);
}
